// Attention_core_20675972563526
// MI455X (gfx1250) — compile-verified, matrix operand order checked by reading, not run on hardware
//
#include <hip/hip_runtime.h>
#include <hip/hip_bf16.h>
#include <math.h>

// Model dims (fixed by the reference)
#define DDIM  128
#define LSEQ  2048
#define NB    8
#define NHEAD 8
#define CHN   512
#define DFF   512   // EXP*D = 4*128

typedef __attribute__((ext_vector_type(16))) __bf16 v16bf;
typedef __attribute__((ext_vector_type(8)))  float  v8f;

// k-index pattern for 16-bit A/B WMMA fragments (ISA 7.12.2):
// vgpr j (0..7), sub-element s in {0,1}: k = 16*(j>=4) + 8*(lane>=16) + 2*(j&3) + s
__device__ __forceinline__ int wmma_kpat(int j, int lane) {
  return ((j & 4) << 2) + ((lane >> 4) << 3) + ((j & 3) << 1);
}

// ---------------------------------------------------------------------------
// Generic bf16-WMMA GEMM:  Y[m,n] = alpha * sum_k X[m,k]*W[n,k]  (+bias)(+gelu)
//                          (+residual R[m,n]) (+instnorm stats atomics)
// Batched over grid.z with (b,h) strides;  M,N multiples of 128; K multiple of 32.
// FLAGS: 1=bias  2=gelu(exact)  4=residual add  8=accumulate sum/sumsq stats
// ---------------------------------------------------------------------------
template <int FLAGS>
__global__ __launch_bounds__(256) void gemm_kernel(
    const float* __restrict__ Xb, const float* __restrict__ Wb,
    const float* __restrict__ bias, const float* __restrict__ Rb,
    float* __restrict__ Yb, float* __restrict__ stats,
    int M, int N, int K, int nh,
    long long sxb, long long sxh, long long swb, long long swh,
    long long syb, long long syh, long long srb, long long srh,
    float alpha)
{
  __shared__ __bf16 Xs[128][33];
  __shared__ __bf16 Ws[128][33];
  __shared__ float redsum[8], redsq[8];

  const int z  = blockIdx.z;
  const int bb = z / nh, hh = z % nh;
  const float* X = Xb + bb * sxb + hh * sxh;
  const float* W = Wb + bb * swb + hh * swh;
  const float* R = (FLAGS & 4) ? (Rb + bb * srb + hh * srh) : nullptr;
  float*       Y = Yb + bb * syb + hh * syh;

  const int m0   = blockIdx.x * 128;
  const int n0   = blockIdx.y * 128;
  const int tid  = threadIdx.x;
  const int wave = tid >> 5, lane = tid & 31;
  const int khalf = lane >> 4, l15 = lane & 15;

  v8f acc[8];
#pragma unroll
  for (int t = 0; t < 8; ++t)
#pragma unroll
    for (int i = 0; i < 8; ++i) acc[t][i] = 0.f;

  for (int kc = 0; kc < K; kc += 32) {
    // cooperative f32 -> bf16 tile staging (coalesced along k)
    for (int i = tid; i < 128 * 32; i += 256) {
      const int r = i >> 5, cc = i & 31;
      Xs[r][cc] = (__bf16)X[(size_t)(m0 + r) * K + kc + cc];
      Ws[r][cc] = (__bf16)W[(size_t)(n0 + r) * K + kc + cc];
    }
    __syncthreads();

    // prefetch next K-tile into cache while WMMAs run (global_prefetch_b8)
    if (kc + 32 < K) {
      const int pr = tid >> 1, pc = (tid & 1) << 4;
      __builtin_prefetch(&X[(size_t)(m0 + pr) * K + kc + 32 + pc], 0, 1);
      __builtin_prefetch(&W[(size_t)(n0 + pr) * K + kc + 32 + pc], 0, 1);
    }

    v16bf afrag;
    const int mA = wave * 16 + l15;
#pragma unroll
    for (int j = 0; j < 8; ++j) {
      const int k = wmma_kpat(j, lane);
      afrag[2 * j]     = Xs[mA][k];
      afrag[2 * j + 1] = Xs[mA][k + 1];
    }
#pragma unroll
    for (int t = 0; t < 8; ++t) {
      v16bf bfrag;
      const int nB = t * 16 + l15;
#pragma unroll
      for (int j = 0; j < 8; ++j) {
        const int k = wmma_kpat(j, lane);
        bfrag[2 * j]     = Ws[nB][k];
        bfrag[2 * j + 1] = Ws[nB][k + 1];
      }
      acc[t] = __builtin_amdgcn_wmma_f32_16x16x32_bf16(
          false, afrag, false, bfrag, (short)0, acc[t], false, false);
    }
    __syncthreads();
  }

  float lsum = 0.f, lsq = 0.f;
#pragma unroll
  for (int t = 0; t < 8; ++t) {
    const int n  = n0 + t * 16 + l15;
    float bv = 0.f;
    if (FLAGS & 1) bv = bias[n];
#pragma unroll
    for (int i = 0; i < 8; ++i) {
      const int m = m0 + wave * 16 + i + khalf * 8;
      float v = alpha * acc[t][i] + bv;
      if (FLAGS & 2) v = 0.5f * v * (1.f + erff(v * 0.70710678118654752f));
      if (FLAGS & 4) v += R[(size_t)m * N + n];
      if (FLAGS & 8) { lsum += v; lsq += v * v; }
      Y[(size_t)m * N + n] = v;
    }
  }

  if (FLAGS & 8) {  // per-(b,h) instnorm statistics
#pragma unroll
    for (int o = 16; o > 0; o >>= 1) {
      lsum += __shfl_xor(lsum, o);
      lsq  += __shfl_xor(lsq, o);
    }
    if (lane == 0) { redsum[wave] = lsum; redsq[wave] = lsq; }
    __syncthreads();
    if (tid == 0) {
      float a = 0.f, b2 = 0.f;
      for (int i = 0; i < 8; ++i) { a += redsum[i]; b2 += redsq[i]; }
      atomicAdd(&stats[2 * z],     a);
      atomicAdd(&stats[2 * z + 1], b2);
    }
  }
}

// ---------------------------------------------------------------------------
// stats -> (mean, rstd) per (b,h);  instnorm eps = 1e-5 over L*128 values
// ---------------------------------------------------------------------------
__global__ void finalize_stats_kernel(const float* __restrict__ stats,
                                      float* __restrict__ mr)
{
  const int i = threadIdx.x;
  const float invn = 1.f / (float)((size_t)LSEQ * DDIM);
  const float s = stats[2 * i], q = stats[2 * i + 1];
  const float mean = s * invn;
  const float var  = q * invn - mean * mean;
  mr[2 * i]     = mean;
  mr[2 * i + 1] = rsqrtf(var + 1e-5f);
}

// ---------------------------------------------------------------------------
// instnorm -> softmax(k) -> ctx = P @ Vt^T  (WMMA) -> mean over heads
// grid: (L/128, B); 256 threads; dynamic LDS: probs[128][128]bf16 + Vt[128][128]bf16
// ---------------------------------------------------------------------------
__global__ __launch_bounds__(256) void softmax_ctx_kernel(
    const float* __restrict__ sbuf, const float* __restrict__ mr,
    const float* __restrict__ vtbuf, float* __restrict__ ctx)
{
  extern __shared__ char smem[];
  __bf16 (*probs)[128] = (__bf16(*)[128])smem;
  __bf16 (*vts)[128]   = (__bf16(*)[128])(smem + 128 * 128 * sizeof(__bf16));

  const int b   = blockIdx.y;
  const int l0  = blockIdx.x * 128;
  const int tid = threadIdx.x;
  const int wave = tid >> 5, lane = tid & 31;
  const int khalf = lane >> 4, l15 = lane & 15;

  v8f acc[8];
#pragma unroll
  for (int t = 0; t < 8; ++t)
#pragma unroll
    for (int i = 0; i < 8; ++i) acc[t][i] = 0.f;

  for (int h = 0; h < NHEAD; ++h) {
    const int z = b * NHEAD + h;
    const float mean = mr[2 * z], rstd = mr[2 * z + 1];

    // stage Vt_h (bf16)
    const float* Vt = vtbuf + (size_t)z * DDIM * DDIM;
    for (int i = tid; i < DDIM * DDIM; i += 256)
      vts[i >> 7][i & 127] = (__bf16)Vt[i];

    // per-row instnorm + softmax -> bf16 probs
    const float* S = sbuf + ((size_t)z * LSEQ + l0) * DDIM;
    for (int rr = 0; rr < 16; ++rr) {
      const int r = wave * 16 + rr;
      const float* srow = S + (size_t)r * DDIM;
      float v[4], mx = -3.4e38f;
#pragma unroll
      for (int j = 0; j < 4; ++j) {
        v[j] = (srow[lane * 4 + j] - mean) * rstd;
        mx = fmaxf(mx, v[j]);
      }
#pragma unroll
      for (int o = 16; o > 0; o >>= 1) mx = fmaxf(mx, __shfl_xor(mx, o));
      float sm = 0.f;
#pragma unroll
      for (int j = 0; j < 4; ++j) { v[j] = expf(v[j] - mx); sm += v[j]; }
#pragma unroll
      for (int o = 16; o > 0; o >>= 1) sm += __shfl_xor(sm, o);
      const float inv = 1.f / sm;
#pragma unroll
      for (int j = 0; j < 4; ++j)
        probs[r][lane * 4 + j] = (__bf16)(v[j] * inv);
    }
    __syncthreads();

    // ctx[l,e] += sum_k P[l,k] * Vt[e,k]
    const int mA = wave * 16 + l15;
    for (int kc = 0; kc < DDIM; kc += 32) {
      v16bf afrag;
#pragma unroll
      for (int j = 0; j < 8; ++j) {
        const int k = kc + wmma_kpat(j, lane);
        afrag[2 * j]     = probs[mA][k];
        afrag[2 * j + 1] = probs[mA][k + 1];
      }
#pragma unroll
      for (int t = 0; t < 8; ++t) {
        v16bf bfrag;
        const int nB = t * 16 + l15;
#pragma unroll
        for (int j = 0; j < 8; ++j) {
          const int k = kc + wmma_kpat(j, lane);
          bfrag[2 * j]     = vts[nB][k];
          bfrag[2 * j + 1] = vts[nB][k + 1];
        }
        acc[t] = __builtin_amdgcn_wmma_f32_16x16x32_bf16(
            false, afrag, false, bfrag, (short)0, acc[t], false, false);
      }
    }
    __syncthreads();
  }

  // write head-mean ctx
#pragma unroll
  for (int t = 0; t < 8; ++t) {
    const int e = t * 16 + l15;
#pragma unroll
    for (int i = 0; i < 8; ++i) {
      const int m = l0 + wave * 16 + i + khalf * 8;
      ctx[((size_t)(b * LSEQ + m)) * DDIM + e] = acc[t][i] * (1.f / NHEAD);
    }
  }
}

// ---------------------------------------------------------------------------
// conv3x3(Cin=128 -> 1) + BN(eval) + ReLU + LayerNorm over row of 128
// strides (sc, sy, sx) select the transpose view of `core`
// grid: (128 rows, B); 128 threads (one per x)
// ---------------------------------------------------------------------------
__global__ __launch_bounds__(128) void conv_cbr_ln_kernel(
    const float* __restrict__ core, const float* __restrict__ Wc,
    const float* __restrict__ bconv,
    const float* __restrict__ bng, const float* __restrict__ bnb,
    const float* __restrict__ bnrm, const float* __restrict__ bnrv,
    const float* __restrict__ ang, const float* __restrict__ anb,
    float* __restrict__ out, int sc, int sy, int sx)
{
  __shared__ float wks[128 * 9];
  __shared__ float red[128];
  const int y = blockIdx.x, b = blockIdx.y, x = threadIdx.x;

  for (int i = x; i < 128 * 9; i += 128) wks[i] = Wc[i];
  __syncthreads();

  const float* cb = core + (size_t)b * DDIM * DDIM * DDIM;
  float acc = 0.f;
  for (int ci = 0; ci < 128; ++ci) {
    const float* wr = &wks[ci * 9];
    const size_t cio = (size_t)ci * sc;
#pragma unroll
    for (int ky = 0; ky < 3; ++ky) {
      const int yy = y + ky - 1;
      if (yy < 0 || yy >= 128) continue;
#pragma unroll
      for (int kx = 0; kx < 3; ++kx) {
        const int xx = x + kx - 1;
        if (xx < 0 || xx >= 128) continue;
        acc += cb[cio + (size_t)yy * sy + (size_t)xx * sx] * wr[ky * 3 + kx];
      }
    }
  }
  float v = acc + bconv[0];
  v = (v - bnrm[0]) * rsqrtf(bnrv[0] + 1e-5f) * bng[0] + bnb[0];
  v = fmaxf(v, 0.f);

  // LayerNorm over the 128-wide row (eps = 1e-6)
  red[x] = v; __syncthreads();
  for (int o = 64; o > 0; o >>= 1) { if (x < o) red[x] += red[x + o]; __syncthreads(); }
  const float mu = red[0] * (1.f / 128.f);
  __syncthreads();
  const float d = v - mu;
  red[x] = d * d; __syncthreads();
  for (int o = 64; o > 0; o >>= 1) { if (x < o) red[x] += red[x + o]; __syncthreads(); }
  const float var = red[0] * (1.f / 128.f);
  out[(size_t)b * DDIM * DDIM + (size_t)y * DDIM + x] =
      d * rsqrtf(var + 1e-6f) * ang[x] + anb[x];
}

// ---------------------------------------------------------------------------
// Row LayerNorm, dim=128, eps=1e-6;  grid: nRows, 128 threads
// ---------------------------------------------------------------------------
__global__ __launch_bounds__(128) void ln_kernel(
    const float* __restrict__ x, const float* __restrict__ g,
    const float* __restrict__ bta, float* __restrict__ y, float eps)
{
  __shared__ float red[128];
  const size_t row = blockIdx.x;
  const int t = threadIdx.x;
  const float v = x[row * 128 + t];
  red[t] = v; __syncthreads();
  for (int o = 64; o > 0; o >>= 1) { if (t < o) red[t] += red[t + o]; __syncthreads(); }
  const float mu = red[0] * (1.f / 128.f);
  __syncthreads();
  const float d = v - mu;
  red[t] = d * d; __syncthreads();
  for (int o = 64; o > 0; o >>= 1) { if (t < o) red[t] += red[t + o]; __syncthreads(); }
  const float var = red[0] * (1.f / 128.f);
  y[row * 128 + t] = d * rsqrtf(var + eps) * g[t] + bta[t];
}

// ---------------------------------------------------------------------------
// Host orchestration
// ---------------------------------------------------------------------------
extern "C" void kernel_launch(void* const* d_in, const int* in_sizes, int n_in,
                              void* d_out, int out_size, void* d_ws, size_t ws_size,
                              hipStream_t stream)
{
  (void)in_sizes; (void)n_in; (void)out_size; (void)ws_size;
  const float* c_in  = (const float*)d_in[0];
  const float* h_in  = (const float*)d_in[1];
  const float* w_in  = (const float*)d_in[2];
  const float* core  = (const float*)d_in[3];
  auto P = [&](int i) { return (const float*)d_in[i]; };
  // params flat order: conv-base 4+br*6 {Wconv,bconv,bn_g,bn_b,bn_rm,bn_rv};
  // per-branch base 22+bi*14 {an_g,an_b,n_g,n_b,Wq,Wk,Wv,Wo,fn_g,fn_b,W1,b1,W2,b2};
  // 64:W_l1 65:b_l1 66:W_l2 67:b_l2

  char* ws = (char*)d_ws;
  size_t off = 0;
  auto carve = [&](size_t elems) -> float* {
    float* p = (float*)(ws + off);
    off += ((elems * 4 + 255) / 256) * 256;
    return p;
  };

  const size_t BL = (size_t)NB * LSEQ;   // 16384
  float* c2     = carve(BL * DDIM);
  float* qln    = carve(BL * DDIM);
  float* coreLn = carve(3ull * NB * DDIM * DDIM);
  float* Qb     = carve((size_t)NB * NHEAD * LSEQ * DDIM);
  float* Kb     = carve((size_t)NB * NHEAD * DDIM * DDIM);
  float* Vtb    = carve((size_t)NB * NHEAD * DDIM * DDIM);
  float* sb     = carve((size_t)NB * NHEAD * LSEQ * DDIM);
  float* stats  = carve((size_t)NB * NHEAD * 2);
  float* mrb    = carve((size_t)NB * NHEAD * 2);
  float* ctx    = carve(BL * DDIM);
  float* Ob     = carve(BL * DDIM);
  float* tb     = carve(BL * DDIM);
  float* ub     = carve(BL * DFF);
  float* O1b    = carve(BL * DDIM);

  float* outf = (float*)d_out;
  float* yfinal[3] = { O1b, outf + BL * CHN, outf + BL * CHN + BL * DDIM };

  // dispatch helper over compile-time epilogue flags
  auto gemm = [&](dim3 grid, const float* X, const float* W, const float* bias,
                  const float* R, float* Y, float* st, int M, int N, int K, int nh,
                  long long sxb, long long sxh, long long swb, long long swh,
                  long long syb, long long syh, long long srb, long long srh,
                  float alpha, int flags) {
    switch (flags) {
      case 0: gemm_kernel<0><<<grid, 256, 0, stream>>>(X, W, bias, R, Y, st, M, N, K, nh,
                  sxb, sxh, swb, swh, syb, syh, srb, srh, alpha); break;
      case 1: gemm_kernel<1><<<grid, 256, 0, stream>>>(X, W, bias, R, Y, st, M, N, K, nh,
                  sxb, sxh, swb, swh, syb, syh, srb, srh, alpha); break;
      case 3: gemm_kernel<3><<<grid, 256, 0, stream>>>(X, W, bias, R, Y, st, M, N, K, nh,
                  sxb, sxh, swb, swh, syb, syh, srb, srh, alpha); break;
      case 4: gemm_kernel<4><<<grid, 256, 0, stream>>>(X, W, bias, R, Y, st, M, N, K, nh,
                  sxb, sxh, swb, swh, syb, syh, srb, srh, alpha); break;
      case 5: gemm_kernel<5><<<grid, 256, 0, stream>>>(X, W, bias, R, Y, st, M, N, K, nh,
                  sxb, sxh, swb, swh, syb, syh, srb, srh, alpha); break;
      case 8: gemm_kernel<8><<<grid, 256, 0, stream>>>(X, W, bias, R, Y, st, M, N, K, nh,
                  sxb, sxh, swb, swh, syb, syh, srb, srh, alpha); break;
      default: break;
    }
  };

  // 1) core branches: conv+BN+ReLU+LN for the three transposes
  {
    dim3 g(DDIM, NB);
    // 'c': channel=i1(D), y=i2(1), x=i0(D*D)
    conv_cbr_ln_kernel<<<g, 128, 0, stream>>>(core, P(4), P(5), P(6), P(7), P(8), P(9),
        P(22), P(23), coreLn + 0ull * NB * DDIM * DDIM, DDIM, 1, DDIM * DDIM);
    // 'h': channel=i2(1), y=i0(D*D), x=i1(D)
    conv_cbr_ln_kernel<<<g, 128, 0, stream>>>(core, P(10), P(11), P(12), P(13), P(14), P(15),
        P(36), P(37), coreLn + 1ull * NB * DDIM * DDIM, 1, DDIM * DDIM, DDIM);
    // 'w': channel=i0(D*D), y=i1(D), x=i2(1)
    conv_cbr_ln_kernel<<<g, 128, 0, stream>>>(core, P(16), P(17), P(18), P(19), P(20), P(21),
        P(50), P(51), coreLn + 2ull * NB * DDIM * DDIM, DDIM * DDIM, DDIM, 1);
  }

  // 2) linear1: c2 = c @ W_l1^T + b_l1   (16384x128, K=512)
  gemm(dim3(BL / 128, 1, 1), c_in, P(64), P(65), nullptr, c2, nullptr,
       (int)BL, DDIM, CHN, 1, 0, 0, 0, 0, 0, 0, 0, 0, 1.f, 1);

  const float* qsrc[3]  = { c2, h_in, w_in };
  const float* resid[3] = { c2, h_in, w_in };
  const float inv_scale = 0.04419417382415922f;  // 1/sqrt(512)

  for (int bi = 0; bi < 3; ++bi) {
    const int pb = 22 + bi * 14;
    const float* corebr = coreLn + (size_t)bi * NB * DDIM * DDIM;

    // q-LN
    ln_kernel<<<(int)BL, 128, 0, stream>>>(qsrc[bi], P(pb + 2), P(pb + 3), qln, 1e-6f);

    // Q[b,h] = qln_b @ Wq_h^T
    gemm(dim3(LSEQ / 128, 1, NB * NHEAD), qln, P(pb + 4), nullptr, nullptr, Qb, nullptr,
         LSEQ, DDIM, DDIM, NHEAD,
         (long long)LSEQ * DDIM, 0, 0, (long long)DDIM * DDIM,
         (long long)NHEAD * LSEQ * DDIM, (long long)LSEQ * DDIM, 0, 0, 1.f, 0);

    // K[b,h] = kv_b @ Wk_h^T
    gemm(dim3(1, 1, NB * NHEAD), corebr, P(pb + 5), nullptr, nullptr, Kb, nullptr,
         DDIM, DDIM, DDIM, NHEAD,
         (long long)DDIM * DDIM, 0, 0, (long long)DDIM * DDIM,
         (long long)NHEAD * DDIM * DDIM, (long long)DDIM * DDIM, 0, 0, 1.f, 0);

    // Vt[b,h][e,k] = sum_d Wv_h[e,d] * kv_b[k,d]
    gemm(dim3(1, 1, NB * NHEAD), P(pb + 6), corebr, nullptr, nullptr, Vtb, nullptr,
         DDIM, DDIM, DDIM, NHEAD,
         0, (long long)DDIM * DDIM, (long long)DDIM * DDIM, 0,
         (long long)NHEAD * DDIM * DDIM, (long long)DDIM * DDIM, 0, 0, 1.f, 0);

    // scores s = Q K^T / sqrt(CH), with instnorm stats
    hipMemsetAsync(stats, 0, (size_t)NB * NHEAD * 2 * sizeof(float), stream);
    gemm(dim3(LSEQ / 128, 1, NB * NHEAD), Qb, Kb, nullptr, nullptr, sb, stats,
         LSEQ, DDIM, DDIM, NHEAD,
         (long long)NHEAD * LSEQ * DDIM, (long long)LSEQ * DDIM,
         (long long)NHEAD * DDIM * DDIM, (long long)DDIM * DDIM,
         (long long)NHEAD * LSEQ * DDIM, (long long)LSEQ * DDIM, 0, 0,
         inv_scale, 8);

    finalize_stats_kernel<<<1, NB * NHEAD, 0, stream>>>(stats, mrb);

    // instnorm -> softmax -> P@Vt^T -> head mean
    softmax_ctx_kernel<<<dim3(LSEQ / 128, NB), 256,
                         2 * DDIM * DDIM * sizeof(__bf16), stream>>>(sb, mrb, Vtb, ctx);

    // O = residual + ctx @ Wo^T
    gemm(dim3(BL / 128, 1, 1), ctx, P(pb + 7), nullptr, resid[bi], Ob, nullptr,
         (int)BL, DDIM, DDIM, 1, 0, 0, 0, 0, 0, 0, 0, 0, 1.f, 4);

    // FFN: t = LN(O); u = gelu(t@W1^T + b1); O = O + u@W2^T + b2
    ln_kernel<<<(int)BL, 128, 0, stream>>>(Ob, P(pb + 8), P(pb + 9), tb, 1e-6f);
    gemm(dim3(BL / 128, DFF / 128, 1), tb, P(pb + 10), P(pb + 11), nullptr, ub, nullptr,
         (int)BL, DFF, DDIM, 1, 0, 0, 0, 0, 0, 0, 0, 0, 1.f, 3);
    gemm(dim3(BL / 128, 1, 1), ub, P(pb + 12), P(pb + 13), Ob, yfinal[bi], nullptr,
         (int)BL, DDIM, DFF, 1, 0, 0, 0, 0, 0, 0, 0, 0, 1.f, 5);
  }

  // linear2: out1 = O1 @ W_l2^T + b_l2   (16384x512, K=128)
  gemm(dim3(BL / 128, CHN / 128, 1), O1b, P(66), P(67), nullptr, outf, nullptr,
       (int)BL, CHN, DDIM, 1, 0, 0, 0, 0, 0, 0, 0, 0, 1.f, 1);
}
